// BondHead2_7395933683794
// MI455X (gfx1250) — matrix operand order repaired and checked by reading, not compiled, not run on hardware
//
#include <hip/hip_runtime.h>
#include <hip/hip_bf16.h>
#include <math.h>

typedef __attribute__((ext_vector_type(16))) _Float16 v16h;
typedef __attribute__((ext_vector_type(8)))  _Float16 half8;
typedef __attribute__((ext_vector_type(4)))  _Float16 half4;
typedef __attribute__((ext_vector_type(8)))  float    v8f;
typedef __attribute__((ext_vector_type(4)))  float    f32x4;

#define HD   64
#define SEQ  384
#define NB   8
#define ROWS (NB*SEQ*SEQ)          // 1,179,648
#define NTILES (ROWS/16)           // 73,728
#define WPB  8                     // waves per block (256 threads, wave32)
#define NBLK 1152                  // 1152*8 = 9216 wave slots -> 8 tiles per wave

__device__ __forceinline__ void wave_lds_fence() {
  asm volatile("s_wait_dscnt 0" ::: "memory");
}

// Branchless GELU. Preferred path uses the gfx1250 hardware tanh transcendental
// (TRANS pipe, co-issues with VALU): ~6 VALU + 1 tanh, |err| ~3e-4 (below the
// f16-WMMA operand noise). Fallback: branchless A&S 7.1.26 erf, |err|<=1.5e-7,
// 1 v_rcp + 1 v_exp + ~10 FMAs, no exec-mask divergence.
__device__ __forceinline__ float gelu_f(float xv) {
#if __has_builtin(__builtin_amdgcn_tanhf)
  float u  = xv * xv;
  float c  = fmaf(0.044715f, u, 1.0f);
  float ar = (xv * 0.7978845608028654f) * c;
  float th = __builtin_amdgcn_tanhf(ar);
  float h  = 0.5f * xv;
  return fmaf(h, th, h);
#else
  float z   = xv * 0.70710678118654752f;
  float az  = fabsf(z);
  float t   = __builtin_amdgcn_rcpf(fmaf(0.3275911f, az, 1.0f));
  float p   = fmaf(1.061405429f, t, -1.453152027f);
  p = fmaf(p, t, 1.421413741f);
  p = fmaf(p, t, -0.284496736f);
  p = fmaf(p, t, 0.254829592f);
  p = p * t;
  float e   = __builtin_amdgcn_exp2f((az * az) * -1.4426950408889634f);
  float era = fmaf(-p, e, 1.0f);
  float er  = copysignf(era, z);
  float h   = 0.5f * xv;
  return fmaf(h, er, h);
#endif
}

__launch_bounds__(256)
__global__ void fused_mlp_kernel(
    const float* __restrict__ x,
    const float* __restrict__ W1, const float* __restrict__ b1,
    const float* __restrict__ W2, const float* __restrict__ b2,
    const float* __restrict__ W3, const float* __restrict__ b3,
    const float* __restrict__ lng_p, const float* __restrict__ lnb_p,
    const float* __restrict__ Wo, const float* __restrict__ bo,
    float* __restrict__ out)
{
  // Weights transposed: sW[l][n*64+k] = W_l[k][n]  (f16)
  __shared__ _Float16 sW[3][HD*HD];     // 24 KB
  __shared__ float    sWoT[5*HD];       // 1.25 KB, transposed: sWoT[v*64+k]=Wo[k][v]
  __shared__ float    sH[WPB][16*HD];   // 32 KB (f32 h3; low half aliased as f16 A stage)

  const int tid  = threadIdx.x;
  const int wid  = tid >> 5;
  const int lane = tid & 31;
  const int r16  = lane & 15;
  const bool hiH = lane >= 16;

  // ---- stage weights (transposed, f16) and Wo^T (f32) into LDS ----
  {
    const float* Wsrc[3] = {W1, W2, W3};
    for (int e = tid; e < 3*HD*HD; e += 256) {
      int l = e / (HD*HD); int r = e % (HD*HD);
      int k = r / HD, n = r % HD;
      sW[l][n*HD + k] = (_Float16)Wsrc[l][r];
    }
    for (int e = tid; e < HD*5; e += 256) {
      int k = e / 5, v = e % 5;
      sWoT[v*HD + k] = Wo[e];
    }
  }
  __syncthreads();

  // ---- per-lane channel parameters (channel n = 16*t + lane%16) ----
  float bs[3][4], lg[4], lb[4];
  #pragma unroll
  for (int t = 0; t < 4; ++t) {
    int n = t*16 + r16;
    bs[0][t] = b1[n]; bs[1][t] = b2[n]; bs[2][t] = b3[n];
    lg[t] = lng_p[n]; lb[t] = lnb_p[n];
  }

  float*    myH = &sH[wid][0];
  _Float16* myA = (_Float16*)myH;   // f16 A stage aliases low 2 KB of the 4 KB region

  for (int tile = blockIdx.x*WPB + wid; tile < NTILES; tile += gridDim.x*WPB) {
    const float* xt = x + (size_t)tile * (16*HD);

    // ---- stage 16x64 f32 tile -> f16 row-major in LDS ----
    #pragma unroll
    for (int i = 0; i < 8; ++i) {
      int u = i*32 + lane;               // 0..255
      int row = u >> 4, seg = u & 15;
      f32x4 val = *(const f32x4*)(xt + row*HD + seg*4);
      half4 hv;
      hv.x = (_Float16)val.x; hv.y = (_Float16)val.y;
      hv.z = (_Float16)val.z; hv.w = (_Float16)val.w;
      *(half4*)(myA + row*HD + seg*4) = hv;
    }
    // speculative prefetch of next tile (dropped if OOB)
    __builtin_prefetch(xt + (size_t)gridDim.x*WPB*16*HD + lane*32, 0, 1);

    // ---- three (GEMM -> GELU -> LN) blocks ----
    #pragma unroll
    for (int l = 0; l < 3; ++l) {
      wave_lds_fence();
      // A fragments (16x32 f16 layout): lane<16 holds K {0..7,16..23}(+32s), lane>=16 {8..15,24..31}
      const _Float16* ap = myA + r16*HD + (hiH ? 8 : 0);
      half8 p0 = *(const half8*)(ap);
      half8 q0 = *(const half8*)(ap + 16);
      half8 p1 = *(const half8*)(ap + 32);
      half8 q1 = *(const half8*)(ap + 48);
      v16h a0 = __builtin_shufflevector(p0, q0, 0,1,2,3,4,5,6,7,8,9,10,11,12,13,14,15);
      v16h a1 = __builtin_shufflevector(p1, q1, 0,1,2,3,4,5,6,7,8,9,10,11,12,13,14,15);

      v8f g[4];
      v8f s1 = {0.f,0.f,0.f,0.f,0.f,0.f,0.f,0.f};
      v8f s2 = {0.f,0.f,0.f,0.f,0.f,0.f,0.f,0.f};

      #pragma unroll
      for (int t = 0; t < 4; ++t) {
        v8f acc;
        #pragma unroll
        for (int e = 0; e < 8; ++e) acc[e] = bs[l][t];   // C initialized to bias

        // B fragments from transposed weights: b[i] = W[32s + off + i + (i>=8 ? 8 : 0)][col], off = hiH?8:0, same K order as A
        const _Float16* wp = &sW[l][(t*16 + r16)*HD + (hiH ? 8 : 0)];
        half8 wb0 = *(const half8*)(wp);
        half8 wb1 = *(const half8*)(wp + 16);
        half8 wb2 = *(const half8*)(wp + 32);
        half8 wb3 = *(const half8*)(wp + 48);
        v16h bf0 = __builtin_shufflevector(wb0, wb1, 0,1,2,3,4,5,6,7,8,9,10,11,12,13,14,15);
        v16h bf1 = __builtin_shufflevector(wb2, wb3, 0,1,2,3,4,5,6,7,8,9,10,11,12,13,14,15);

        acc = __builtin_amdgcn_wmma_f32_16x16x32_f16(false, a0, false, bf0, (short)0, acc, false, false);
        acc = __builtin_amdgcn_wmma_f32_16x16x32_f16(false, a1, false, bf1, (short)0, acc, false, false);

        #pragma unroll
        for (int e = 0; e < 8; ++e) {
          float ge = gelu_f(acc[e]);
          g[t][e] = ge;
          s1[e] += ge;
          s2[e] += ge * ge;
        }
      }

      // LayerNorm stats: butterfly within each 16-lane half (each row lives on one half)
      #pragma unroll
      for (int e = 0; e < 8; ++e) {
        float a = s1[e], b = s2[e];
        #pragma unroll
        for (int d = 1; d < 16; d <<= 1) {
          a += __shfl_xor(a, d, 32);
          b += __shfl_xor(b, d, 32);
        }
        float mu  = a * (1.0f/64.0f);
        float var = b * (1.0f/64.0f) - mu*mu;
        s1[e] = mu;
        s2[e] = rsqrtf(var + 1e-5f);
      }

      // normalize + scale/shift; stage for next layer (f16) or final proj (f32)
      #pragma unroll
      for (int t = 0; t < 4; ++t) {
        #pragma unroll
        for (int e = 0; e < 8; ++e) {
          float yv = (g[t][e] - s1[e]) * s2[e] * lg[t] + lb[t];
          int row = e + (hiH ? 8 : 0);
          int n   = t*16 + r16;
          if (l < 2) myA[row*HD + n] = (_Float16)yv;
          else       myH[row*HD + n] = yv;
        }
      }
    }

    wave_lds_fence();
    // ---- final projection: (16x64 f32) @ (64x5) + bo, fp32 VALU, b128 LDS reads ----
    #pragma unroll
    for (int it = 0; it < 3; ++it) {
      int item = it*32 + lane;           // 80 items = 16 rows x 5 vocab
      if (item < 80) {
        int row = item / 5, v = item % 5;
        float sum = bo[v];
        const float* hrow = myH + row*HD;
        const float* wv   = &sWoT[v*HD];
        #pragma unroll
        for (int k0 = 0; k0 < HD; k0 += 4) {
          f32x4 hv = *(const f32x4*)(hrow + k0);
          f32x4 wq = *(const f32x4*)(wv + k0);
          sum = fmaf(hv.x, wq.x, sum);
          sum = fmaf(hv.y, wq.y, sum);
          sum = fmaf(hv.z, wq.z, sum);
          sum = fmaf(hv.w, wq.w, sum);
        }
        out[(size_t)(tile*16 + row)*5 + v] = sum;
      }
    }
  }
}

// In-place symmetrization: out = 0.5*(out + out^T) over the two seq axes.
// Each (i<j) pair is owned by exactly one thread -> race-free & deterministic.
__global__ void sym_kernel(float* __restrict__ out)
{
  size_t idx = (size_t)blockIdx.x * 256 + threadIdx.x;
  const size_t total = (size_t)NB * SEQ * SEQ * 5;
  if (idx >= total) return;
  int v = (int)(idx % 5);
  size_t r = idx / 5;
  int j = (int)(r % SEQ); r /= SEQ;
  int i = (int)(r % SEQ);
  int b = (int)(r / SEQ);
  if (i < j) {
    size_t cidx = ((((size_t)b * SEQ + j) * SEQ) + i) * 5 + v;
    float a = out[idx], c = out[cidx];
    float m = 0.5f * (a + c);
    out[idx]  = m;
    out[cidx] = m;
  }
}

extern "C" void kernel_launch(void* const* d_in, const int* in_sizes, int n_in,
                              void* d_out, int out_size, void* d_ws, size_t ws_size,
                              hipStream_t stream) {
  const float* x   = (const float*)d_in[0];
  const float* W1  = (const float*)d_in[1];
  const float* b1  = (const float*)d_in[2];
  const float* W2  = (const float*)d_in[3];
  const float* b2  = (const float*)d_in[4];
  const float* W3  = (const float*)d_in[5];
  const float* b3  = (const float*)d_in[6];
  const float* lng = (const float*)d_in[7];
  const float* lnb = (const float*)d_in[8];
  const float* Wo  = (const float*)d_in[9];
  const float* bo  = (const float*)d_in[10];
  float* out = (float*)d_out;

  fused_mlp_kernel<<<NBLK, 256, 0, stream>>>(x, W1, b1, W2, b2, W3, b3,
                                             lng, lnb, Wo, bo, out);

  const size_t total = (size_t)NB * SEQ * SEQ * 5;
  int blocks = (int)((total + 255) / 256);
  sym_kernel<<<blocks, 256, 0, stream>>>(out);
}
